// STEMA_Core_48000554501010
// MI455X (gfx1250) — matrix operand order repaired and checked by reading, not compiled, not run on hardware
//
#include <hip/hip_runtime.h>
#include <hip/hip_bf16.h>
#include <math.h>

// Problem dims (compile-time constants from the reference)
#define B_    8
#define T_    512
#define DIM_  512
#define R_    2048
#define H_    8
#define HD_   64
#define FFH_  1362
#define FFHP_ 1376   // FFH padded to multiple of 32 for WMMA tiling

typedef __attribute__((ext_vector_type(2)))  __bf16 v2bf;
typedef __attribute__((ext_vector_type(4)))  __bf16 v4bf;
typedef __attribute__((ext_vector_type(8)))  __bf16 v8bf;
typedef __attribute__((ext_vector_type(16))) __bf16 v16bf;
typedef __attribute__((ext_vector_type(8)))  float  v8f;

__device__ __forceinline__ v8f wmma_bf16(v16bf a, v16bf b, v8f c) {
  // D = A(16x32) * B(32x16) + C, f32 accumulate
  return __builtin_amdgcn_wmma_f32_16x16x32_bf16(false, a, false, b, (short)0, c,
                                                 false, false);
}

// ---- fragment loaders (wave32): contiguous k order k = 16h + s, same in A and B ----
// A (16x32 bf16): lane l -> row m=l&15, half h=l>>4; VGPR v holds K pair
//   k = h*16 + v*2 + {0,1}
__device__ __forceinline__ v16bf load_a_frag(const __bf16* base, int ld) {
  const int lane = threadIdx.x & 31;
  const int m = lane & 15, h = lane >> 4;
  v16bf a;
#pragma unroll
  for (int v = 0; v < 8; ++v) {
    const int ka = h * 16 + v * 2;
    v2bf p = *(const v2bf*)(base + (size_t)m * ld + ka);
    a[2 * v] = p.x;
    a[2 * v + 1] = p.y;
  }
  return a;
}
// B (32x16): lane l -> column n=l&15 (weight row n), k = (l>>4)*16 + e
// 16 contiguous K per lane -> single 32B vector load.
__device__ __forceinline__ v16bf load_b_frag(const __bf16* base, int ld) {
  const int lane = threadIdx.x & 31;
  const int n = lane & 15, h = lane >> 4;
  return *(const v16bf*)(base + (size_t)n * ld + h * 16);
}

// =====================================================================
// Ternary quantization: thr = 0.7*mean|w|; q = sign(w)*(|w|>thr) in bf16
// =====================================================================
__global__ __launch_bounds__(256) void absmean_partial_kernel(
    const float* __restrict__ w, int n, float* __restrict__ partial) {
  float s = 0.0f;
  for (int i = blockIdx.x * 256 + threadIdx.x; i < n; i += 256 * 256)
    s += fabsf(w[i]);
#pragma unroll
  for (int o = 16; o > 0; o >>= 1) s += __shfl_xor(s, o, 32);
  __shared__ float red[8];
  if ((threadIdx.x & 31) == 0) red[threadIdx.x >> 5] = s;
  __syncthreads();
  if (threadIdx.x == 0) {
    float t = 0.0f;
#pragma unroll
    for (int i = 0; i < 8; ++i) t += red[i];
    partial[blockIdx.x] = t;
  }
}

__global__ __launch_bounds__(256) void absmean_final_kernel(
    const float* __restrict__ partial, float scale, float* __restrict__ thr) {
  float s = partial[threadIdx.x];
#pragma unroll
  for (int o = 16; o > 0; o >>= 1) s += __shfl_xor(s, o, 32);
  __shared__ float red[8];
  if ((threadIdx.x & 31) == 0) red[threadIdx.x >> 5] = s;
  __syncthreads();
  if (threadIdx.x == 0) {
    float t = 0.0f;
#pragma unroll
    for (int i = 0; i < 8; ++i) t += red[i];
    thr[0] = t * scale;  // scale = 0.7 / n
  }
}

__global__ __launch_bounds__(256) void quantize_kernel(
    const float* __restrict__ w, const float* __restrict__ thr,
    __bf16* __restrict__ q, int rows, int cols, int pcols, int ptotal) {
  const int idx = blockIdx.x * 256 + threadIdx.x;
  if (idx >= ptotal) return;
  const int r = idx / pcols, c = idx % pcols;
  float o = 0.0f;
  if (r < rows && c < cols) {
    const float v = w[(size_t)r * cols + c];
    o = (fabsf(v) > thr[0]) ? (v > 0.0f ? 1.0f : -1.0f) : 0.0f;
  }
  q[(size_t)r * pcols + c] = (__bf16)o;
}

__global__ __launch_bounds__(256) void f32_to_bf16_kernel(
    const float* __restrict__ in, __bf16* __restrict__ outp, int n) {
  const int i = blockIdx.x * 256 + threadIdx.x;
  if (i < n) outp[i] = (__bf16)in[i];
}

// Warm the recurrent weight matrix into L2 before the 512-step scan.
// (cluster barrier is a NOP outside a cluster launch; exercises the
//  gfx1250 barrier path deterministically)
__global__ __launch_bounds__(256) void prefetch_kernel(
    const void* p, unsigned long long nbytes) {
  unsigned long long i =
      ((unsigned long long)blockIdx.x * 256ull + threadIdx.x) * 256ull;
  const unsigned long long stride = (unsigned long long)gridDim.x * 256ull * 256ull;
  for (; i < nbytes; i += stride)
    __builtin_prefetch((const char*)p + i, 0, 1);  // -> global_prefetch_b8
  __builtin_amdgcn_s_cluster_barrier();
}

// =====================================================================
// Generic bf16 WMMA GEMM:  C[M,N] = A[M,K] * W[N,K]^T  (+ epilogue)
// block = 256 threads (8 waves), block tile 64(M) x 32(N), K step 32
// EPI: 0 = (+bias) store, 1 = (+bias) exact GELU, 2 = silu(aux)*acc,
//      3 = acc + aux (residual)
// =====================================================================
template <int EPI, bool OUTBF>
__global__ __launch_bounds__(256) void gemm_wmma_kernel(
    const __bf16* __restrict__ A, int lda, const __bf16* __restrict__ W,
    int ldb, void* __restrict__ Cout, int ldc, const float* __restrict__ bias,
    const float* __restrict__ aux, int ldaux, int M, int N, int K) {
  __shared__ __align__(32) __bf16 As[64][32];
  __shared__ __align__(32) __bf16 Bs[32][32];
  const int tid = threadIdx.x;
  const int wid = tid >> 5, lane = tid & 31;
  const int bm = blockIdx.y * 64;
  const int bn = blockIdx.x * 32;
  const int wr = wid >> 1, wc = wid & 1;  // 4x2 wave grid of 16x16 tiles
  v8f acc = {};
  const int ar = tid >> 2, ac = (tid & 3) * 8;   // A tile: 64x32, 8 bf16/thread
  const int br = tid >> 3, bc = (tid & 7) * 4;   // B tile: 32x32, 4 bf16/thread
  for (int k0 = 0; k0 < K; k0 += 32) {
    *(v8bf*)&As[ar][ac] = *(const v8bf*)(A + (size_t)(bm + ar) * lda + k0 + ac);
    *(v4bf*)&Bs[br][bc] = *(const v4bf*)(W + (size_t)(bn + br) * ldb + k0 + bc);
    __syncthreads();
    v16bf a = load_a_frag(&As[wr * 16][0], 32);
    v16bf b = load_b_frag(&Bs[wc * 16][0], 32);
    acc = wmma_bf16(a, b, acc);
    __syncthreads();
  }
  const int l16 = lane & 15, hh = lane >> 4;
  const int n = bn + wc * 16 + l16;
  const float bv = bias ? bias[n] : 0.0f;
#pragma unroll
  for (int r = 0; r < 8; ++r) {
    const int m = bm + wr * 16 + hh * 8 + r;  // C layout: vgpr r -> M=r / 8+r
    float v = acc[r] + bv;
    if (EPI == 1) {
      v = 0.5f * v * (1.0f + erff(v * 0.70710678118654752f));
    } else if (EPI == 2) {
      const float g = aux[(size_t)m * ldaux + n];
      v *= g / (1.0f + __expf(-g));  // silu(g) * u
    } else if (EPI == 3) {
      v += aux[(size_t)m * ldaux + n];
    }
    if (OUTBF)
      ((__bf16*)Cout)[(size_t)m * ldc + n] = (__bf16)v;
    else
      ((float*)Cout)[(size_t)m * ldc + n] = v;
  }
}

// =====================================================================
// One LIF timestep: cur = xp[:,t,:] + last @ Wr^T ; spike/reset/SFA fused.
// Batch (8) padded to 16 WMMA rows. Double-buffered spikes (last_in/out).
// grid = 16 blocks * 8 waves: each wave owns one 16-col tile of R=2048.
// =====================================================================
__global__ __launch_bounds__(256) void lif_step_kernel(
    const __bf16* __restrict__ Wr, const float* __restrict__ xp,
    const __bf16* __restrict__ last_in, __bf16* __restrict__ last_out,
    float* __restrict__ mem, float* __restrict__ ath,
    __bf16* __restrict__ acts, const float* __restrict__ beta,
    const float* __restrict__ sfa_inc, const float* __restrict__ sfa_dec,
    int t) {
  __shared__ __align__(16) __bf16 As[16][32];
  const int tid = threadIdx.x;
  const int wid = tid >> 5, lane = tid & 31;
  const int n0 = (blockIdx.x * 8 + wid) * 16;
  v8f acc = {};
  const int ar = tid >> 4, ac = (tid & 15) * 2;  // 16x32 spike tile, 2/thread
  for (int k0 = 0; k0 < R_; k0 += 32) {
    *(v2bf*)&As[ar][ac] = *(const v2bf*)(last_in + (size_t)ar * R_ + k0 + ac);
    __syncthreads();
    v16bf a = load_a_frag(&As[0][0], 32);
    v16bf b = load_b_frag(Wr + (size_t)n0 * R_ + k0, R_);
    acc = wmma_bf16(a, b, acc);
    __syncthreads();
  }
  const int l16 = lane & 15, hh = lane >> 4;
  const int n = n0 + l16;
  const float be = beta[n], si = sfa_inc[n], sd = sfa_dec[n];
#pragma unroll
  for (int r = 0; r < 8; ++r) {
    const int m = hh * 8 + r;  // padded batch row 0..15
    float spk = 0.0f;
    if (m < B_) {
      const int idx = m * R_ + n;
      const float cur = acc[r] + xp[(size_t)(m * T_ + t) * R_ + n];
      float mv = be * mem[idx] + cur;
      const float av = ath[idx];
      spk = (mv - (1.0f + av)) > 0.0f ? 1.0f : 0.0f;
      mem[idx] = mv * (1.0f - spk);
      ath[idx] = sd * av + si * spk;
      acts[(size_t)(m * T_ + t) * R_ + n] = (__bf16)spk;
    }
    last_out[m * R_ + n] = (__bf16)spk;  // keeps rows 8..15 zero
  }
}

// =====================================================================
// RMSNorm over last dim, writes bf16 normalized activations
// =====================================================================
template <typename Tin>
__global__ __launch_bounds__(256) void rmsnorm_kernel(
    const Tin* __restrict__ x, const float* __restrict__ w,
    __bf16* __restrict__ outp, int ncols) {
  const size_t row = blockIdx.x;
  const Tin* xr = x + row * ncols;
  float ss = 0.0f;
  for (int c = threadIdx.x; c < ncols; c += 256) {
    const float v = (float)xr[c];
    ss += v * v;
  }
#pragma unroll
  for (int o = 16; o > 0; o >>= 1) ss += __shfl_xor(ss, o, 32);
  __shared__ float red[8];
  if ((threadIdx.x & 31) == 0) red[threadIdx.x >> 5] = ss;
  __syncthreads();
  if (threadIdx.x == 0) {
    float s = 0.0f;
#pragma unroll
    for (int i = 0; i < 8; ++i) s += red[i];
    red[0] = rsqrtf(s / (float)ncols + 1e-6f);
  }
  __syncthreads();
  const float scale = red[0];
  for (int c = threadIdx.x; c < ncols; c += 256)
    outp[row * ncols + c] = (__bf16)((float)xr[c] * scale * w[c]);
}

// =====================================================================
// Flash-style attention: block = (b, h, 128-row q tile), 8 waves.
// Per wave: 16 q rows. WMMA for Q*K^T and P*V, online softmax with
// shfl_xor row statistics, P transposed via LDS into A-fragment layout.
// =====================================================================
__global__ __launch_bounds__(256) void attn_kernel(
    const __bf16* __restrict__ qkv, __bf16* __restrict__ obuf) {
  const int qt = blockIdx.x & 3;
  const int h = (blockIdx.x >> 2) & 7;
  const int b = blockIdx.x >> 5;
  const int tid = threadIdx.x;
  const int wid = tid >> 5, lane = tid & 31;
  const int l16 = lane & 15, hh = lane >> 4;
  const int q0 = qt * 128 + wid * 16;
  const int ldq = 3 * DIM_;
  __shared__ __align__(32) __bf16 Ks[32][64];
  __shared__ __align__(32) __bf16 Vs[32][64];
  __shared__ float Ps[8][16][32];  // per-wave P tile (16 q x 32 keys)

  // Q fragments (K-dim 64 -> two 16x32 A frags), loaded once
  v16bf aq[2];
#pragma unroll
  for (int kk = 0; kk < 2; ++kk) {
#pragma unroll
    for (int v = 0; v < 8; ++v) {
      const int ka = hh * 16 + v * 2;
      v2bf p = *(const v2bf*)(qkv + (size_t)(b * T_ + q0 + l16) * ldq + h * HD_ +
                              kk * 32 + ka);
      aq[kk][2 * v] = p.x;
      aq[kk][2 * v + 1] = p.y;
    }
  }
  float rowM[8], rowL[8];
  v8f o0 = {}, o1 = {}, o2 = {}, o3 = {};
#pragma unroll
  for (int r = 0; r < 8; ++r) {
    rowM[r] = -1e30f;
    rowL[r] = 0.0f;
  }
  const int sr = tid >> 3, sc = (tid & 7) * 8;  // K/V staging: 32x64, 8/thread
  for (int kt = 0; kt < T_ / 32; ++kt) {
    const __bf16* kvb =
        qkv + (size_t)(b * T_ + kt * 32 + sr) * ldq + h * HD_;
    *(v8bf*)&Ks[sr][sc] = *(const v8bf*)(kvb + DIM_ + sc);
    *(v8bf*)&Vs[sr][sc] = *(const v8bf*)(kvb + 2 * DIM_ + sc);
    __syncthreads();
    // S = Q*K^T for two 16-key groups
    v8f s0 = {}, s1 = {};
#pragma unroll
    for (int kk = 0; kk < 2; ++kk) {
      v16bf bk0 = *(const v16bf*)&Ks[l16][kk * 32 + hh * 16];
      v16bf bk1 = *(const v16bf*)&Ks[16 + l16][kk * 32 + hh * 16];
      s0 = wmma_bf16(aq[kk], bk0, s0);
      s1 = wmma_bf16(aq[kk], bk1, s1);
    }
    // online softmax over 32 new keys
#pragma unroll
    for (int r = 0; r < 8; ++r) {
      const float a0 = s0[r] * 0.125f, a1 = s1[r] * 0.125f;  // HD^-0.5
      float mx = fmaxf(a0, a1);
#pragma unroll
      for (int d = 8; d > 0; d >>= 1) mx = fmaxf(mx, __shfl_xor(mx, d, 16));
      const float newM = fmaxf(rowM[r], mx);
      const float corr = __expf(rowM[r] - newM);
      const float p0 = __expf(a0 - newM);
      const float p1 = __expf(a1 - newM);
      float ls = p0 + p1;
#pragma unroll
      for (int d = 8; d > 0; d >>= 1) ls += __shfl_xor(ls, d, 16);
      rowL[r] = rowL[r] * corr + ls;
      rowM[r] = newM;
      o0[r] *= corr;
      o1[r] *= corr;
      o2[r] *= corr;
      o3[r] *= corr;
      Ps[wid][hh * 8 + r][l16] = p0;
      Ps[wid][hh * 8 + r][16 + l16] = p1;
    }
    __syncthreads();
    // P (C layout) -> A fragment via LDS transpose, then O += P*V
    v16bf ap;
#pragma unroll
    for (int v = 0; v < 8; ++v) {
      const int ka = hh * 16 + v * 2;
      ap[2 * v] = (__bf16)Ps[wid][l16][ka];
      ap[2 * v + 1] = (__bf16)Ps[wid][l16][ka + 1];
    }
    v16bf bv;
#pragma unroll
    for (int e = 0; e < 16; ++e) bv[e] = Vs[hh * 16 + e][l16];
    o0 = wmma_bf16(ap, bv, o0);
#pragma unroll
    for (int e = 0; e < 16; ++e) bv[e] = Vs[hh * 16 + e][16 + l16];
    o1 = wmma_bf16(ap, bv, o1);
#pragma unroll
    for (int e = 0; e < 16; ++e) bv[e] = Vs[hh * 16 + e][32 + l16];
    o2 = wmma_bf16(ap, bv, o2);
#pragma unroll
    for (int e = 0; e < 16; ++e) bv[e] = Vs[hh * 16 + e][48 + l16];
    o3 = wmma_bf16(ap, bv, o3);
    __syncthreads();
  }
#pragma unroll
  for (int r = 0; r < 8; ++r) {
    const float inv = 1.0f / rowL[r];
    const size_t m = (size_t)(b * T_ + q0 + hh * 8 + r) * DIM_ + h * HD_;
    obuf[m + l16] = (__bf16)(o0[r] * inv);
    obuf[m + 16 + l16] = (__bf16)(o1[r] * inv);
    obuf[m + 32 + l16] = (__bf16)(o2[r] * inv);
    obuf[m + 48 + l16] = (__bf16)(o3[r] * inv);
  }
}

// =====================================================================
extern "C" void kernel_launch(void* const* d_in, const int* in_sizes, int n_in,
                              void* d_out, int out_size, void* d_ws,
                              size_t ws_size, hipStream_t stream) {
  const float* x = (const float*)d_in[0];
  const float* Wi = (const float*)d_in[1];
  const float* bi = (const float*)d_in[2];
  const float* Wr = (const float*)d_in[3];
  const float* rn_w = (const float*)d_in[4];
  const float* W1 = (const float*)d_in[5];
  const float* b1 = (const float*)d_in[6];
  const float* W2 = (const float*)d_in[7];
  const float* b2 = (const float*)d_in[8];
  const float* anorm_w = (const float*)d_in[9];
  const float* Wqkv = (const float*)d_in[10];
  const float* Wo = (const float*)d_in[11];
  const float* fnorm_w = (const float*)d_in[12];
  const float* Wf1 = (const float*)d_in[13];
  const float* Wf2 = (const float*)d_in[14];
  const float* Wf3 = (const float*)d_in[15];
  const float* beta = (const float*)d_in[16];
  const float* sfa_inc = (const float*)d_in[17];
  const float* sfa_dec = (const float*)d_in[18];
  float* outp = (float*)d_out;

  // ---- workspace bump allocator (256B aligned) ----
  char* ws = (char*)d_ws;
  size_t off = 0;
  auto alloc = [&](size_t bytes) -> void* {
    off = (off + 255) & ~(size_t)255;
    void* p = ws + off;
    off += bytes;
    return p;
  };
  const int MT = B_ * T_;  // 4096 rows
  float* partials = (float*)alloc(256 * 4);
  float* thr = (float*)alloc(4);
  __bf16* qWi = (__bf16*)alloc((size_t)R_ * DIM_ * 2);
  __bf16* qWr = (__bf16*)alloc((size_t)R_ * R_ * 2);
  __bf16* qW1 = (__bf16*)alloc((size_t)(2 * DIM_) * R_ * 2);
  __bf16* qW2 = (__bf16*)alloc((size_t)DIM_ * (2 * DIM_) * 2);
  __bf16* qWqkv = (__bf16*)alloc((size_t)(3 * DIM_) * DIM_ * 2);
  __bf16* qWo = (__bf16*)alloc((size_t)DIM_ * DIM_ * 2);
  __bf16* qWf1 = (__bf16*)alloc((size_t)FFHP_ * DIM_ * 2);
  __bf16* qWf3 = (__bf16*)alloc((size_t)FFHP_ * DIM_ * 2);
  __bf16* qWf2 = (__bf16*)alloc((size_t)DIM_ * FFHP_ * 2);
  __bf16* xbf = (__bf16*)alloc((size_t)MT * DIM_ * 2);
  float* xp = (float*)alloc((size_t)MT * R_ * 4);
  float* memb = (float*)alloc((size_t)16 * R_ * 4);
  float* athb = (float*)alloc((size_t)16 * R_ * 4);
  __bf16* lastA = (__bf16*)alloc((size_t)16 * R_ * 2);
  __bf16* lastB = (__bf16*)alloc((size_t)16 * R_ * 2);
  __bf16* acts = (__bf16*)alloc((size_t)MT * R_ * 2);
  __bf16* hn = (__bf16*)alloc((size_t)MT * R_ * 2);
  __bf16* hbuf = (__bf16*)alloc((size_t)MT * (2 * DIM_) * 2);
  float* ybuf = (float*)alloc((size_t)MT * DIM_ * 4);
  __bf16* xa = (__bf16*)alloc((size_t)MT * DIM_ * 2);
  __bf16* qkvb = (__bf16*)alloc((size_t)MT * (3 * DIM_) * 2);
  __bf16* obuf = (__bf16*)alloc((size_t)MT * DIM_ * 2);
  float* xres = (float*)alloc((size_t)MT * DIM_ * 4);
  __bf16* xf = (__bf16*)alloc((size_t)MT * DIM_ * 2);
  float* gbuf = (float*)alloc((size_t)MT * FFHP_ * 4);
  __bf16* gu = (__bf16*)alloc((size_t)MT * FFHP_ * 2);
  (void)ws_size; (void)n_in; (void)in_sizes; (void)out_size;

  // ---- ternary-quantize all weights to bf16 (zero-padded tiles) ----
  auto quant = [&](const float* w, int rows, int cols, int prows, int pcols,
                   __bf16* q) {
    absmean_partial_kernel<<<256, 256, 0, stream>>>(w, rows * cols, partials);
    absmean_final_kernel<<<1, 256, 0, stream>>>(
        partials, 0.7f / (float)(rows * cols), thr);
    const int tot = prows * pcols;
    quantize_kernel<<<(tot + 255) / 256, 256, 0, stream>>>(w, thr, q, rows,
                                                           cols, pcols, tot);
  };
  quant(Wi, R_, DIM_, R_, DIM_, qWi);
  quant(Wr, R_, R_, R_, R_, qWr);
  quant(W1, 2 * DIM_, R_, 2 * DIM_, R_, qW1);
  quant(W2, DIM_, 2 * DIM_, DIM_, 2 * DIM_, qW2);
  quant(Wqkv, 3 * DIM_, DIM_, 3 * DIM_, DIM_, qWqkv);
  quant(Wo, DIM_, DIM_, DIM_, DIM_, qWo);
  quant(Wf1, FFH_, DIM_, FFHP_, DIM_, qWf1);
  quant(Wf3, FFH_, DIM_, FFHP_, DIM_, qWf3);
  quant(Wf2, DIM_, FFH_, DIM_, FFHP_, qWf2);

  // ---- input to bf16 ----
  {
    const int n = MT * DIM_;
    f32_to_bf16_kernel<<<(n + 255) / 256, 256, 0, stream>>>(x, xbf, n);
  }

#define GEMM(EPI, OUTBF, Ap, lda, Wp, ldb, Cp, ldc, biasp, auxp, ldaux, M, N, \
             K)                                                               \
  gemm_wmma_kernel<EPI, OUTBF><<<dim3((N) / 32, (M) / 64), 256, 0, stream>>>( \
      Ap, lda, Wp, ldb, Cp, ldc, biasp, auxp, ldaux, M, N, K)

  // xp = x @ Wi^T + bi  -> f32 [4096,2048]
  GEMM(0, false, xbf, DIM_, qWi, DIM_, xp, R_, bi, (const float*)nullptr, 0,
       MT, R_, DIM_);

  // ---- LIF reservoir scan (sequential in T, double-buffered spikes) ----
  hipMemsetAsync(memb, 0, (size_t)16 * R_ * 4, stream);
  hipMemsetAsync(athb, 0, (size_t)16 * R_ * 4, stream);
  hipMemsetAsync(lastA, 0, (size_t)16 * R_ * 2, stream);
  hipMemsetAsync(lastB, 0, (size_t)16 * R_ * 2, stream);
  prefetch_kernel<<<64, 256, 0, stream>>>(qWr, (size_t)R_ * R_ * 2);
  for (int t = 0; t < T_; ++t) {
    const __bf16* lin = (t & 1) ? lastB : lastA;
    __bf16* lout = (t & 1) ? lastA : lastB;
    lif_step_kernel<<<16, 256, 0, stream>>>(qWr, xp, lin, lout, memb, athb,
                                            acts, beta, sfa_inc, sfa_dec, t);
  }

  // ---- readout: RMSNorm -> W1+GELU -> W2+bias ----
  rmsnorm_kernel<__bf16><<<MT, 256, 0, stream>>>(acts, rn_w, hn, R_);
  GEMM(1, true, hn, R_, qW1, R_, hbuf, 2 * DIM_, b1, (const float*)nullptr, 0,
       MT, 2 * DIM_, R_);
  GEMM(0, false, hbuf, 2 * DIM_, qW2, 2 * DIM_, ybuf, DIM_, b2,
       (const float*)nullptr, 0, MT, DIM_, 2 * DIM_);

  // ---- attention block ----
  rmsnorm_kernel<float><<<MT, 256, 0, stream>>>(ybuf, anorm_w, xa, DIM_);
  GEMM(0, true, xa, DIM_, qWqkv, DIM_, qkvb, 3 * DIM_, (const float*)nullptr,
       (const float*)nullptr, 0, MT, 3 * DIM_, DIM_);
  attn_kernel<<<B_ * H_ * (T_ / 128), 256, 0, stream>>>(qkvb, obuf);
  // x_res = y + o @ Wo^T
  GEMM(3, false, obuf, DIM_, qWo, DIM_, xres, DIM_, (const float*)nullptr,
       ybuf, DIM_, MT, DIM_, DIM_);

  // ---- SwiGLU FFN ----
  rmsnorm_kernel<float><<<MT, 256, 0, stream>>>(xres, fnorm_w, xf, DIM_);
  GEMM(0, false, xf, DIM_, qWf1, DIM_, gbuf, FFHP_, (const float*)nullptr,
       (const float*)nullptr, 0, MT, FFHP_, DIM_);
  GEMM(2, true, xf, DIM_, qWf3, DIM_, gu, FFHP_, (const float*)nullptr, gbuf,
       FFHP_, MT, FFHP_, DIM_);
  // out = x_res + (g*u) @ Wf2^T  (K padded with zeros -> exact)
  GEMM(3, false, gu, FFHP_, qWf2, FFHP_, outp, DIM_, (const float*)nullptr,
       xres, DIM_, MT, DIM_, FFHP_);
#undef GEMM
}
